// Transition_13675175870789
// MI455X (gfx1250) — matrix operand order repaired and checked by reading, not compiled, not run on hardware
//
#include <hip/hip_runtime.h>
#include <hip/hip_bf16.h>
#include <stddef.h>
#include <stdint.h>

// ---------------------------------------------------------------------------
// Transition block: LN -> Linear(1024->4096)+ReLU -> Linear(4096->1024)
// M = 4*2048 = 8192 rows, C = 1024, C_INT = 4096.
// Compute-bound (AI ~1370 flop/byte) -> bf16 WMMA, f32 accumulation,
// LDS double-buffered via CDNA5 async global->LDS copies (ASYNCcnt).
// ---------------------------------------------------------------------------

#define M_ROWS 8192
#define C_DIM  1024
#define CI_DIM 4096

typedef __attribute__((ext_vector_type(16))) __bf16 v16bf;
typedef __attribute__((ext_vector_type(8)))  float  v8f;

// ---------------------------------------------------------------------------
// CDNA5 async global->LDS copy, 16B per lane.  Tracked on ASYNCcnt.
// LDS byte address from VGPR %0; global addr = VGPR pair %1 (GV mode).
// ---------------------------------------------------------------------------
__device__ __forceinline__ void async_b128(uint32_t lds_byte_off, const void* gsrc) {
    asm volatile("global_load_async_to_lds_b128 %0, %1, off"
                 :: "v"(lds_byte_off), "v"(gsrc)
                 : "memory");
}

__device__ __forceinline__ uint32_t lds_addr(const void* p) {
    // Flat pointers to LDS carry the LDS byte offset in the low 32 bits.
    return (uint32_t)(uintptr_t)p;
}

// ---------------------------------------------------------------------------
// Kernel 1: LayerNorm (f32 in) -> bf16 out.  One wave32 per row of 1024.
// ---------------------------------------------------------------------------
__global__ void ln_to_bf16_kernel(const float* __restrict__ x,
                                  const float* __restrict__ gamma,
                                  const float* __restrict__ beta,
                                  __bf16* __restrict__ xb) {
    int row  = blockIdx.x * 8 + (threadIdx.x >> 5);
    int lane = threadIdx.x & 31;
    const float* xr = x + (size_t)row * C_DIM;

    float v[32];
    float s = 0.f;
#pragma unroll
    for (int t = 0; t < 32; ++t) {
        v[t] = xr[lane + 32 * t];
        s += v[t];
    }
#pragma unroll
    for (int m = 16; m; m >>= 1) s += __shfl_xor(s, m, 32);
    float mu = s * (1.0f / C_DIM);

    float s2 = 0.f;
#pragma unroll
    for (int t = 0; t < 32; ++t) {
        float d = v[t] - mu;
        s2 += d * d;
    }
#pragma unroll
    for (int m = 16; m; m >>= 1) s2 += __shfl_xor(s2, m, 32);
    float rs = rsqrtf(s2 * (1.0f / C_DIM) + 1e-5f);

    __bf16* xo = xb + (size_t)row * C_DIM;
#pragma unroll
    for (int t = 0; t < 32; ++t) {
        int c = lane + 32 * t;
        xo[c] = (__bf16)((v[t] - mu) * rs * gamma[c] + beta[c]);
    }
}

// ---------------------------------------------------------------------------
// Kernel 2: pack row-major f32 weight (K x N) into bf16 WMMA B-fragment
// layout: per (32x16) KxN tile, 512 elems, element (k_local, n_local) at
//   lane = (k_local/16)*16 + n_local, i = k_local%16
// tiles ordered n-tile-major, k-tile-minor (contiguous K stream per n-tile).
// ---------------------------------------------------------------------------
__global__ void pack_w_kernel(const float* __restrict__ w,
                              __bf16* __restrict__ wp,
                              int K, int N) {
    int o    = blockIdx.x * 256 + threadIdx.x;   // packed element index
    int tile = o >> 9;                           // / 512
    int lane = (o >> 4) & 31;
    int i    = o & 15;
    int ktiles = K >> 5;
    int nt = tile / ktiles;
    int kt = tile - nt * ktiles;
    int n  = (nt << 4) + (lane & 15);
    int k  = (kt << 5) + ((lane >> 4) << 4) + i;
    wp[o] = (__bf16)w[(size_t)k * N + n];
}

// ---------------------------------------------------------------------------
// Kernel 3: WMMA GEMM with LDS double buffering fed by async copies.
// Block = 256 threads = 8 waves (2 x 4).  Block tile 128x128, K-step 32.
// Wave tile 64(M) x 32(N) = 4x2 f32 16x16 accumulators.
// LDS layout == WMMA fragment layout: per tile, lane*32B holds that lane's
// 16 bf16 fragment, so GEMM reads are single aligned 32B LDS loads.
// Each thread stages (m-tile|n-tile == its wave id, its lane) via
// 4 x global_load_async_to_lds_b128 per K-step.
// ---------------------------------------------------------------------------
template <int K, int N, bool RELU_BF16>
__global__ void gemm_wmma_lds_kernel(const __bf16* __restrict__ A,
                                     const __bf16* __restrict__ Bp,
                                     const float*  __restrict__ bias,
                                     __bf16* __restrict__ Obf,
                                     float*  __restrict__ Of) {
    constexpr int WM = 4, WN = 2;
    constexpr int ktiles = K >> 5;

    __shared__ __align__(32) __bf16 sA[2][8 * 512];   // 8 m-tiles x 512 elems
    __shared__ __align__(32) __bf16 sB[2][8 * 512];   // 8 n-tiles x 512 elems

    const int tid   = threadIdx.x;
    const int wave  = tid >> 5;
    const int lane  = tid & 31;
    const int waveM = wave >> 2;   // 0..1
    const int waveN = wave & 3;    // 0..3
    const int m_block = blockIdx.y * 128;
    const int n_block = blockIdx.x * 128;
    const int half = lane >> 4;
    const int l15  = lane & 15;

    // Copy sources: this thread fills m-tile == wave and n-tile == wave.
    // A fragment (row-major source): row = m_block + wave*16 + l15,
    //   piece0 = K half*16..+7, piece1 = K half*16+8..+7  (16B each).
    const __bf16* a_src = A + (size_t)(m_block + wave * 16 + l15) * K + half * 16;
    // B packed source: n-tile chunk is already in fragment order.
    const __bf16* b_src = Bp + ((size_t)((n_block >> 4) + wave) * ktiles) * 512
                             + lane * 16;

    const uint32_t a_dst[2] = { lds_addr(&sA[0][wave * 512 + lane * 16]),
                                lds_addr(&sA[1][wave * 512 + lane * 16]) };
    const uint32_t b_dst[2] = { lds_addr(&sB[0][wave * 512 + lane * 16]),
                                lds_addr(&sB[1][wave * 512 + lane * 16]) };

    auto issue_copies = [&](int kt, int buf) {
        const __bf16* as = a_src + kt * 32;
        const __bf16* bs = b_src + (size_t)kt * 512;
        async_b128(a_dst[buf],      as);
        async_b128(a_dst[buf] + 16, as + 8);
        async_b128(b_dst[buf],      bs);
        async_b128(b_dst[buf] + 16, bs + 8);
    };

    v8f acc[WM][WN] = {};

    issue_copies(0, 0);

    for (int kt = 0; kt < ktiles; ++kt) {
        const int buf = kt & 1;
        if (kt + 1 < ktiles) {
            issue_copies(kt + 1, buf ^ 1);
            // 4 copies for the next buffer remain in flight; the 4 for the
            // current buffer (issued earlier) must have landed.
            asm volatile("s_wait_asynccnt 0x4" ::: "memory");
        } else {
            asm volatile("s_wait_asynccnt 0x0" ::: "memory");
        }
        __syncthreads();

        v16bf a[WM], b[WN];
#pragma unroll
        for (int i = 0; i < WM; ++i)
            a[i] = *(const v16bf*)&sA[buf][(waveM * 4 + i) * 512 + lane * 16];
#pragma unroll
        for (int j = 0; j < WN; ++j)
            b[j] = *(const v16bf*)&sB[buf][(waveN * 2 + j) * 512 + lane * 16];

#pragma unroll
        for (int i = 0; i < WM; ++i)
#pragma unroll
            for (int j = 0; j < WN; ++j)
                acc[i][j] = __builtin_amdgcn_wmma_f32_16x16x32_bf16(
                    false, a[i], false, b[j], (short)0, acc[i][j],
                    false, false);

        __syncthreads();   // everyone done reading buf before it is refilled
    }

    // Epilogue. C/D layout: lanes 0-15 -> N=lane, M=r; lanes 16-31 -> N=lane-16, M=r+8.
#pragma unroll
    for (int i = 0; i < WM; ++i) {
        int mrow0 = m_block + (waveM * WM + i) * 16 + half * 8;
#pragma unroll
        for (int j = 0; j < WN; ++j) {
            int ncol = n_block + (waveN * WN + j) * 16 + l15;
            float bs = bias[ncol];
#pragma unroll
            for (int r = 0; r < 8; ++r) {
                float val = acc[i][j][r] + bs;
                if constexpr (RELU_BF16) {
                    val = fmaxf(val, 0.f);
                    Obf[(size_t)(mrow0 + r) * N + ncol] = (__bf16)val;
                } else {
                    Of[(size_t)(mrow0 + r) * N + ncol] = val;
                }
            }
        }
    }
}

// ---------------------------------------------------------------------------
// Launch
// ---------------------------------------------------------------------------
extern "C" void kernel_launch(void* const* d_in, const int* in_sizes, int n_in,
                              void* d_out, int out_size, void* d_ws, size_t ws_size,
                              hipStream_t stream) {
    const float* act      = (const float*)d_in[0];
    // d_in[1] = mask (unused by reference forward)
    const float* ln_scale = (const float*)d_in[2];
    const float* ln_bias  = (const float*)d_in[3];
    const float* w1       = (const float*)d_in[4];
    const float* b1       = (const float*)d_in[5];
    const float* w2       = (const float*)d_in[6];
    const float* b2       = (const float*)d_in[7];
    float* out = (float*)d_out;

    char* ws = (char*)d_ws;
    __bf16* xb  = (__bf16*)(ws);                        // 8192x1024 bf16 (16 MB)
    __bf16* w1p = (__bf16*)(ws + (size_t)16777216);     // packed 1024x4096 (8 MB)
    __bf16* w2p = (__bf16*)(ws + (size_t)25165824);     // packed 4096x1024 (8 MB)
    __bf16* h   = (__bf16*)(ws + (size_t)33554432);     // 8192x4096 bf16 (64 MB)

    // 1) LayerNorm -> bf16 (8 rows per 256-thread block)
    ln_to_bf16_kernel<<<M_ROWS / 8, 256, 0, stream>>>(act, ln_scale, ln_bias, xb);

    // 2) Pack weights into B-fragment layout (bf16)
    pack_w_kernel<<<(C_DIM * CI_DIM) / 256, 256, 0, stream>>>(w1, w1p, C_DIM, CI_DIM);
    pack_w_kernel<<<(CI_DIM * C_DIM) / 256, 256, 0, stream>>>(w2, w2p, CI_DIM, C_DIM);

    // 3) GEMM1: (8192x1024)x(1024x4096), +b1, ReLU -> h (bf16)
    gemm_wmma_lds_kernel<C_DIM, CI_DIM, true>
        <<<dim3(CI_DIM / 128, M_ROWS / 128), 256, 0, stream>>>(xb, w1p, b1, h, nullptr);

    // 4) GEMM2: (8192x4096)x(4096x1024), +b2 -> out (f32)
    gemm_wmma_lds_kernel<CI_DIM, C_DIM, false>
        <<<dim3(C_DIM / 128, M_ROWS / 128), 256, 0, stream>>>(h, w2p, b2, nullptr, out);
}
